// SAHead_45054206935084
// MI455X (gfx1250) — matrix operand order repaired and checked by reading, not compiled, not run on hardware
//
#include <hip/hip_runtime.h>

typedef _Float16 v16h __attribute__((ext_vector_type(16)));
typedef _Float16 v8h  __attribute__((ext_vector_type(8)));
typedef float    v8f  __attribute__((ext_vector_type(8)));

#define B_    2
#define CIN   64
#define CENC  96
#define DK    64
#define DV    64
#define HW    96
#define LSP   (HW * HW)      // 9216 spatial positions

// flash-attention tiling
#define BM   128             // query rows per block (8 waves x 16)
#define BN   64              // key block
#define KSTR 72              // LDS row stride (halves): 36-bank step, conflict-free b128
#define VSTR 72
#define PSTR 72

#define LOG2E 1.44269504089f

__device__ __forceinline__ v16h mk16(v8h lo, v8h hi) {
  v16h r;
#pragma unroll
  for (int i = 0; i < 8; ++i) { r[i] = lo[i]; r[i + 8] = hi[i]; }
  return r;
}

// D = A(16x32 f16) * B(32x16 f16) + C(16x16 f32)
__device__ __forceinline__ v8f wmma16(v16h a, v16h b, v8f c) {
  return __builtin_amdgcn_wmma_f32_16x16x32_f16(false, a, false, b,
                                                (short)0, c, false, false);
}

// async global->LDS copy of 16 bytes (per lane), tracked by ASYNCcnt
__device__ __forceinline__ void async_b128(const void* gptr, const void* lptr) {
  const unsigned long long ga = (unsigned long long)(uintptr_t)gptr;
  const unsigned la = (unsigned)(uintptr_t)lptr;   // low 32 bits = LDS offset
  asm volatile("global_load_async_to_lds_b128 %0, %1, off"
               :: "v"(la), "v"(ga) : "memory");
}

// ---------------------------------------------------------------------------
// Projection Q/K: 1x1 conv (dk x Cenc) + bias, per-position L2 normalize.
// Q is additionally pre-scaled by log2(e) so attention uses raw v_exp_f32.
// Layout: (B, L, dk) row-major f16.
// ---------------------------------------------------------------------------
__global__ __launch_bounds__(128) void proj_qk_kernel(
    const float* __restrict__ xenc,
    const float* __restrict__ Wq, const float* __restrict__ bq,
    const float* __restrict__ Wk, const float* __restrict__ bk,
    _Float16* __restrict__ Qw, _Float16* __restrict__ Kw)
{
  __shared__ float xs[CENC][128];
  const int b    = blockIdx.y;
  const int pos0 = blockIdx.x * 128;
  const int t    = threadIdx.x;

  const float* xp = xenc + (size_t)b * CENC * LSP + pos0;
  for (int c = 0; c < CENC; ++c) xs[c][t] = xp[(size_t)c * LSP + t];
  __syncthreads();

  const int pos = pos0 + t;
  float acc[DK];

  // ---- Q (pre-scaled by log2 e) ----
#pragma unroll
  for (int k = 0; k < DK; ++k) acc[k] = bq[k];
  for (int c = 0; c < CENC; ++c) {
    const float xv = xs[c][t];
#pragma unroll
    for (int k = 0; k < DK; ++k) acc[k] = fmaf(Wq[k * CENC + c], xv, acc[k]);
  }
  {
    float ss = 0.f;
#pragma unroll
    for (int k = 0; k < DK; ++k) ss += acc[k] * acc[k];
    const float inv = LOG2E / fmaxf(sqrtf(ss), 1e-6f);
    _Float16* qp = Qw + ((size_t)b * LSP + pos) * DK;
#pragma unroll
    for (int kk = 0; kk < 8; ++kk) {
      v8h tmp;
#pragma unroll
      for (int j = 0; j < 8; ++j) tmp[j] = (_Float16)(acc[kk * 8 + j] * inv);
      *(v8h*)(qp + kk * 8) = tmp;
    }
  }

  // ---- K ----
#pragma unroll
  for (int k = 0; k < DK; ++k) acc[k] = bk[k];
  for (int c = 0; c < CENC; ++c) {
    const float xv = xs[c][t];
#pragma unroll
    for (int k = 0; k < DK; ++k) acc[k] = fmaf(Wk[k * CENC + c], xv, acc[k]);
  }
  {
    float ss = 0.f;
#pragma unroll
    for (int k = 0; k < DK; ++k) ss += acc[k] * acc[k];
    const float inv = 1.0f / fmaxf(sqrtf(ss), 1e-6f);
    _Float16* kp = Kw + ((size_t)b * LSP + pos) * DK;
#pragma unroll
    for (int kk = 0; kk < 8; ++kk) {
      v8h tmp;
#pragma unroll
      for (int j = 0; j < 8; ++j) tmp[j] = (_Float16)(acc[kk * 8 + j] * inv);
      *(v8h*)(kp + kk * 8) = tmp;
    }
  }
}

// ---------------------------------------------------------------------------
// Projection V: 1x1 conv (dv x Cin) + bias, f16 channel-major (B, dv, L).
// ---------------------------------------------------------------------------
__global__ __launch_bounds__(128) void proj_v_kernel(
    const float* __restrict__ x,
    const float* __restrict__ Wv, const float* __restrict__ bv,
    _Float16* __restrict__ Vw)
{
  __shared__ float xs[CIN][128];
  const int b    = blockIdx.y;
  const int pos0 = blockIdx.x * 128;
  const int t    = threadIdx.x;

  const float* xp = x + (size_t)b * CIN * LSP + pos0;
  for (int c = 0; c < CIN; ++c) xs[c][t] = xp[(size_t)c * LSP + t];
  __syncthreads();

  const int pos = pos0 + t;
  float acc[DV];
#pragma unroll
  for (int v = 0; v < DV; ++v) acc[v] = bv[v];
  for (int c = 0; c < CIN; ++c) {
    const float xv = xs[c][t];
#pragma unroll
    for (int v = 0; v < DV; ++v) acc[v] = fmaf(Wv[v * CIN + c], xv, acc[v]);
  }
#pragma unroll
  for (int v = 0; v < DV; ++v)
    Vw[((size_t)b * DV + v) * LSP + pos] = (_Float16)acc[v];
}

// ---------------------------------------------------------------------------
// Fused attention. Scores are cosine similarities in [-1,1] (Q,K unit-norm),
// so exp() cannot overflow: no running max, no O-rescale, deferred row-sum.
// K/V tiles stream into double-buffered LDS with global_load_async_to_lds,
// prefetched one iteration ahead (ASYNCcnt pipelining).
// Per 64-key block: 8 WMMAs (S = Q.K^T) + 8 WMMAs (O += P.V).
// ---------------------------------------------------------------------------
__global__ __launch_bounds__(256) void attn_kernel(
    const _Float16* __restrict__ Qw, const _Float16* __restrict__ Kw,
    const _Float16* __restrict__ Vw, float* __restrict__ out)
{
  __shared__ __attribute__((aligned(16))) _Float16 Kl[2 * BN * KSTR];
  __shared__ __attribute__((aligned(16))) _Float16 Vl[2 * DV * VSTR];
  __shared__ __attribute__((aligned(16))) _Float16 Pl[8 * 16 * PSTR];

  const int b    = blockIdx.y;
  const int t    = threadIdx.x;
  const int w    = t >> 5;
  const int lane = t & 31;
  const int half = lane >> 4;
  const int l16  = lane & 15;
  const int m0   = blockIdx.x * BM + w * 16;

  // per-thread staging slice: 32B of K and 32B of V^T per tile
  const int srow = t >> 2;           // 0..63 (key row / v channel)
  const int scc  = (t & 3) * 16;     // 0,16,32,48
  const _Float16* kg0 = Kw + ((size_t)b * LSP + srow) * DK + scc;
  const _Float16* vg0 = Vw + ((size_t)b * DV + srow) * LSP + scc;

  // Q A-fragments (two 32-wide channel slices), resident for the whole pass.
  v16h qa[2];
  {
    const _Float16* qp = Qw + ((size_t)b * LSP + m0 + l16) * DK;
#pragma unroll
    for (int s = 0; s < 2; ++s) {
      const int kb = 32 * s + 8 * half;
      qa[s] = mk16(*(const v8h*)(qp + kb), *(const v8h*)(qp + kb + 16));
    }
  }

  float l_r[8];            // per-lane partial softmax denominators
  v8f o[4];
#pragma unroll
  for (int r = 0; r < 8; ++r) l_r[r] = 0.f;
#pragma unroll
  for (int tt = 0; tt < 4; ++tt) {
#pragma unroll
    for (int r = 0; r < 8; ++r) o[tt][r] = 0.f;
  }

  _Float16* Pw = Pl + w * 16 * PSTR;

  // issue async fill of buffer 0 (tile j0 = 0)
  {
    _Float16* kd = Kl + srow * KSTR + scc;
    _Float16* vd = Vl + srow * VSTR + scc;
    async_b128(kg0,      kd);
    async_b128(kg0 + 8,  kd + 8);
    async_b128(vg0,      vd);
    async_b128(vg0 + 8,  vd + 8);
  }

  const int NIT = LSP / BN;
  for (int it = 0; it < NIT; ++it) {
    const int cur = it & 1;
    // our async loads for the current tile are done; barrier makes everyone's
    // loads visible and confirms all waves finished reading the spare buffer.
    asm volatile("s_wait_asynccnt 0" ::: "memory");
    __syncthreads();

    // prefetch next tile into the spare buffer (covered by this iteration)
    if (it + 1 < NIT) {
      const int nj = (it + 1) * BN;
      _Float16* kd = Kl + (cur ^ 1) * (BN * KSTR) + srow * KSTR + scc;
      _Float16* vd = Vl + (cur ^ 1) * (DV * VSTR) + srow * VSTR + scc;
      async_b128(kg0 + (size_t)nj * DK,     kd);
      async_b128(kg0 + (size_t)nj * DK + 8, kd + 8);
      async_b128(vg0 + nj,                  vd);
      async_b128(vg0 + nj + 8,              vd + 8);
    }

    const _Float16* Kb = Kl + cur * (BN * KSTR);
    const _Float16* Vb = Vl + cur * (DV * VSTR);

    // ---- S = Q . K_blk^T : four 16x16 tiles; P = exp2(S) staged to LDS
#pragma unroll
    for (int nt = 0; nt < 4; ++nt) {
      v8f s;
#pragma unroll
      for (int r = 0; r < 8; ++r) s[r] = 0.f;
#pragma unroll
      for (int ks = 0; ks < 2; ++ks) {
        const _Float16* kp = Kb + (nt * 16 + l16) * KSTR + ks * 32 + 8 * half;
        s = wmma16(qa[ks], mk16(*(const v8h*)kp, *(const v8h*)(kp + 16)), s);
      }
#pragma unroll
      for (int r = 0; r < 8; ++r) {
        // |s| <= log2(e): raw v_exp_f32, no range fixup needed
        const float p = __builtin_amdgcn_exp2f(s[r]);
        l_r[r] += p;
        Pw[(8 * half + r) * PSTR + nt * 16 + l16] = (_Float16)p;
      }
    }

    // wave-private LDS RAW: LDS is in-order per wave, just drain DScnt
    asm volatile("s_wait_dscnt 0" ::: "memory");

    // ---- O += P . V_blk : two key-slices x four v-channel tiles
#pragma unroll
    for (int ks = 0; ks < 2; ++ks) {
      const _Float16* pp = Pw + l16 * PSTR + ks * 32 + 8 * half;
      const v16h pa = mk16(*(const v8h*)pp, *(const v8h*)(pp + 16));
#pragma unroll
      for (int tt = 0; tt < 4; ++tt) {
        const _Float16* vp = Vb + (tt * 16 + l16) * VSTR + ks * 32 + 8 * half;
        o[tt] = wmma16(pa, mk16(*(const v8h*)vp, *(const v8h*)(vp + 16)), o[tt]);
      }
    }
  }

  // ---- epilogue: one-time row-sum reduction across the 16 lanes of a row
  float invl[8];
#pragma unroll
  for (int r = 0; r < 8; ++r) {
    float rs = l_r[r];
#pragma unroll
    for (int off = 1; off < 16; off <<= 1)
      rs += __shfl_xor(rs, off, 32);
    invl[r] = 1.0f / rs;
  }
#pragma unroll
  for (int tt = 0; tt < 4; ++tt) {
    const int vch = tt * 16 + l16;
    float* op = out + ((size_t)b * DV + vch) * LSP + m0 + 8 * half;
    float4 lo4, hi4;
    lo4.x = o[tt][0] * invl[0]; lo4.y = o[tt][1] * invl[1];
    lo4.z = o[tt][2] * invl[2]; lo4.w = o[tt][3] * invl[3];
    hi4.x = o[tt][4] * invl[4]; hi4.y = o[tt][5] * invl[5];
    hi4.z = o[tt][6] * invl[6]; hi4.w = o[tt][7] * invl[7];
    *(float4*)(op)     = lo4;
    *(float4*)(op + 4) = hi4;
  }
}

// ---------------------------------------------------------------------------
extern "C" void kernel_launch(void* const* d_in, const int* in_sizes, int n_in,
                              void* d_out, int out_size, void* d_ws, size_t ws_size,
                              hipStream_t stream) {
  const float* x    = (const float*)d_in[0];
  const float* xenc = (const float*)d_in[1];
  const float* Wq   = (const float*)d_in[2];
  const float* bq   = (const float*)d_in[3];
  const float* Wk   = (const float*)d_in[4];
  const float* bk   = (const float*)d_in[5];
  const float* Wv   = (const float*)d_in[6];
  const float* bv   = (const float*)d_in[7];
  float* out = (float*)d_out;

  // workspace: Q (B*L*dk f16) | K (B*L*dk f16) | V^T (B*dv*L f16)  ~= 7.1 MB
  _Float16* Qw = (_Float16*)d_ws;
  _Float16* Kw = Qw + (size_t)B_ * LSP * DK;
  _Float16* Vw = Kw + (size_t)B_ * LSP * DK;

  proj_qk_kernel<<<dim3(LSP / 128, B_), 128, 0, stream>>>(xenc, Wq, bq, Wk, bk, Qw, Kw);
  proj_v_kernel <<<dim3(LSP / 128, B_), 128, 0, stream>>>(x, Wv, bv, Vw);
  attn_kernel   <<<dim3(LSP / BM, B_), 256, 0, stream>>>(Qw, Kw, Vw, out);
}
